// STGCNBlock_32057635898067
// MI455X (gfx1250) — matrix operand order repaired and checked by reading, not compiled, not run on hardware
//
#include <hip/hip_runtime.h>
#include <hip/hip_bf16.h>

#define BATCH 8
#define NNODE 512
#define CIN   64
#define HCH   64
#define OUTC  64
#define LSEQ  256
#define KW    3
#define BN_EPS 1e-5f

typedef __attribute__((ext_vector_type(16))) __bf16 v16bf;
typedef __attribute__((ext_vector_type(8)))  float  v8f;

// ---------------------------------------------------------------------------
// WMMA wrapper (wave32, 16x16x32 bf16 -> f32 accumulate)
// ---------------------------------------------------------------------------
__device__ __forceinline__ v8f wmma_bf16(v16bf a, v16bf b, v8f c) {
  return __builtin_amdgcn_wmma_f32_16x16x32_bf16(
      /*neg_a=*/false, a, /*neg_b=*/false, b,
      /*c_mod=*/(short)0, c, /*reuse_a=*/false, /*reuse_b=*/false);
}

// Pack two f32 into a (lo,hi) bf16 pair in one u32.
__device__ __forceinline__ unsigned pack_bf16(float lo, float hi) {
  union { __bf16 b[2]; unsigned u; } r;
  r.b[0] = (__bf16)lo;
  r.b[1] = (__bf16)hi;
  return r.u;
}

// ---------------------------------------------------------------------------
// A fragment 16x32 (MxK) from bf16 row-major LDS [row][K], stride lda (bf16).
// One contiguous 32B run at +half*32B => 2 x ds_load_b128, no movs.
// ---------------------------------------------------------------------------
__device__ __forceinline__ v16bf lda_packed(const __bf16* A, int lda, int row0,
                                            int kb, int lane) {
  int m = lane & 15, half = lane >> 4;
  const uint4* ap = (const uint4*)(A + (row0 + m) * lda + kb + half * 16);
  union { v16bf v; uint4 q[2]; } r;
  r.q[0] = ap[0];   // K pairs 0..3 of the lane's 16-run
  r.q[1] = ap[1];   // K pairs 4..7
  return r.v;
}

// ---------------------------------------------------------------------------
// Fragment-ready B tiles: Bt[kt][ct][lane][8 u32]. Lane = half*16 + n holds
// its 8 WMMA K-pairs contiguously (32B, 32B-aligned) => 2 x ds_load_b128
// directly into the operand quad. bidx() maps a (K-pair, column) to its slot.
// ---------------------------------------------------------------------------
__device__ __forceinline__ int bidx(int k2, int col, int nct) {
  int kt = k2 >> 4, kw = k2 & 15;
  int half = kw >> 3, p = kw & 7;
  int ct = col >> 4, n = col & 15;
  return (((kt * nct + ct) * 32) + half * 16 + n) * 8 + p;
}

__device__ __forceinline__ v16bf ldb_tile(const unsigned* Bt, int nct, int kt,
                                          int ct, int lane) {
  const uint4* p = (const uint4*)(Bt + (size_t)(((kt * nct + ct) * 32) + lane) * 8);
  union { v16bf v; uint4 q[2]; } r;
  r.q[0] = p[0];
  r.q[1] = p[1];
  return r.v;
}

// ---------------------------------------------------------------------------
// K1: temporal conv1. Per (b,n): C(64x256) = w1(64x192) * im2col(x)(192x256).
// B staged from global as fragment-ready packed pairs (rows kk=c*3+k,
// bounds-checked l+k-1). 8 waves, each a 16x128 strip of C.
// ---------------------------------------------------------------------------
__global__ __launch_bounds__(256) void k_tconv1(const float* __restrict__ x,
                                                const float* __restrict__ w1,
                                                const float* __restrict__ b1,
                                                float* __restrict__ out1) {
  extern __shared__ char smem[];
  unsigned* Bim = (unsigned*)smem;                       // 6*16*32*8 u32
  __bf16*   wA  = (__bf16*)(smem + 96 * LSEQ * 4);       // [64][192]
  int tid = threadIdx.x;
  int bn  = blockIdx.x;
  const float* xsl = x + (size_t)bn * (CIN * LSEQ);

  for (int i = tid; i < 96 * LSEQ; i += 256) {           // packed im2col
    int k2 = i >> 8, col = i & 255;
    int kk = 2 * k2;
    int c0 = kk / 3,       km0 = kk - 3 * c0;
    int c1 = (kk + 1) / 3, km1 = (kk + 1) - 3 * c1;
    int l0 = col + km0 - 1, l1 = col + km1 - 1;
    float v0 = (l0 >= 0 && l0 < LSEQ) ? xsl[c0 * LSEQ + l0] : 0.f;
    float v1 = (l1 >= 0 && l1 < LSEQ) ? xsl[c1 * LSEQ + l1] : 0.f;
    Bim[bidx(k2, col, 16)] = pack_bf16(v0, v1);
  }
  {                                                      // weights: flat pairs
    unsigned* wAu = (unsigned*)wA;
    const float2* w1v = (const float2*)w1;
    for (int i = tid; i < HCH * 192 / 2; i += 256) {
      float2 t = w1v[i];
      wAu[i] = pack_bf16(t.x, t.y);
    }
  }
  __syncthreads();

  int lane = tid & 31, wave = tid >> 5;
  int row0 = (wave & 3) * 16;
  int ct0 = (wave >> 2) * 8;
  v8f acc[8] = {};
  for (int kt = 0; kt < 6; ++kt) {
    v16bf a = lda_packed(wA, 192, row0, kt * 32, lane);
#pragma unroll
    for (int t = 0; t < 8; ++t) {
      v16bf b = ldb_tile(Bim, 16, kt, ct0 + t, lane);
      acc[t] = wmma_bf16(a, b, acc[t]);
    }
  }
  int n = lane & 15, half = lane >> 4;
  float* osl = out1 + (size_t)bn * (HCH * LSEQ);
#pragma unroll
  for (int t = 0; t < 8; ++t) {
    int col = (ct0 + t) * 16 + n;
#pragma unroll
    for (int r = 0; r < 8; ++r) {
      int h = row0 + r + half * 8;
      osl[h * LSEQ + col] = acc[t][r] + b1[h];
    }
  }
}

// ---------------------------------------------------------------------------
// K2: xw[d,l] = sum_h gw[h,d] * out1[h,l]   (A = gw^T), K=64
// ---------------------------------------------------------------------------
__global__ __launch_bounds__(256) void k_gw(const float* __restrict__ out1,
                                            const float* __restrict__ gw,
                                            float* __restrict__ xw) {
  extern __shared__ char smem[];
  unsigned* Bp = (unsigned*)smem;                        // 2*16*32*8 u32
  __bf16*   wA = (__bf16*)(smem + 32 * LSEQ * 4);        // [64][64] = gw^T
  int tid = threadIdx.x;
  int bn  = blockIdx.x;
  const float* isl = out1 + (size_t)bn * (HCH * LSEQ);

  for (int i = tid; i < 32 * LSEQ; i += 256) {
    int k2 = i >> 8, col = i & 255;
    Bp[bidx(k2, col, 16)] =
        pack_bf16(isl[(2 * k2) * LSEQ + col], isl[(2 * k2 + 1) * LSEQ + col]);
  }
  {                                                      // gw^T, packed along h
    unsigned* wAu = (unsigned*)wA;
    for (int i = tid; i < HCH * HCH / 2; i += 256) {
      int d = i >> 5, h2 = i & 31;
      wAu[i] = pack_bf16(gw[(2 * h2) * HCH + d], gw[(2 * h2 + 1) * HCH + d]);
    }
  }
  __syncthreads();

  int lane = tid & 31, wave = tid >> 5;
  int row0 = (wave & 3) * 16;
  int ct0 = (wave >> 2) * 8;
  v8f acc[8] = {};
  for (int kt = 0; kt < 2; ++kt) {
    v16bf a = lda_packed(wA, HCH, row0, kt * 32, lane);
#pragma unroll
    for (int t = 0; t < 8; ++t) {
      v16bf b = ldb_tile(Bp, 16, kt, ct0 + t, lane);
      acc[t] = wmma_bf16(a, b, acc[t]);
    }
  }
  int n = lane & 15, half = lane >> 4;
  float* osl = xw + (size_t)bn * (HCH * LSEQ);
#pragma unroll
  for (int t = 0; t < 8; ++t) {
    int col = (ct0 + t) * 16 + n;
#pragma unroll
    for (int r = 0; r < 8; ++r)
      osl[(row0 + r + half * 8) * LSEQ + col] = acc[t][r];
  }
}

// ---------------------------------------------------------------------------
// K3: per-batch GEMM  out[n,(d,l)] = sum_m adj[n,m]*xw[m,(d,l)] + gb[d]
// M=512, K=512, Ncols=16384. WG tile 64x128, K staged 32/iter. Per-batch
// working set ~33 MB << 192 MB L2, so cross-tile B re-reads stay on-chip.
// ---------------------------------------------------------------------------
__global__ __launch_bounds__(256) void k_graph(const float* __restrict__ adj,
                                               const float* __restrict__ xw,
                                               const float* __restrict__ gb,
                                               float* __restrict__ out) {
  extern __shared__ char smem[];
  unsigned* As = (unsigned*)smem;                        // [64][16] u32 (=32 bf16)
  unsigned* Bp = (unsigned*)(smem + 64 * 16 * 4);        // 1*8*32*8 u32
  int tid = threadIdx.x;
  int b = blockIdx.z, ny = blockIdx.y, cx = blockIdx.x;
  int colbase = cx * 128;
  const float* adjb = adj + (size_t)b * NNODE * NNODE + (size_t)ny * 64 * NNODE;
  const float* xwb  = xw + (size_t)b * NNODE * (HCH * LSEQ);
  int lane = tid & 31, wave = tid >> 5;
  int row0 = (wave & 3) * 16;
  int ct0 = (wave >> 2) * 4;
  v8f acc[4] = {};
  for (int m0 = 0; m0 < NNODE; m0 += 32) {
    for (int i = tid; i < 64 * 16; i += 256) {           // A tile, packed pairs
      int r = i >> 4, c2 = i & 15;
      As[i] = pack_bf16(adjb[r * NNODE + m0 + 2 * c2],
                        adjb[r * NNODE + m0 + 2 * c2 + 1]);
    }
    for (int i = tid; i < 16 * 128; i += 256) {          // B tile, frag-ready
      int k2 = i >> 7, col = i & 127;
      Bp[bidx(k2, col, 8)] =
          pack_bf16(xwb[(size_t)(m0 + 2 * k2) * (HCH * LSEQ) + colbase + col],
                    xwb[(size_t)(m0 + 2 * k2 + 1) * (HCH * LSEQ) + colbase + col]);
    }
    __syncthreads();
    v16bf a = lda_packed((const __bf16*)As, 32, row0, 0, lane);
#pragma unroll
    for (int t = 0; t < 4; ++t) {
      v16bf bf = ldb_tile(Bp, 8, 0, ct0 + t, lane);
      acc[t] = wmma_bf16(a, bf, acc[t]);
    }
    __syncthreads();
  }
  int n = lane & 15, half = lane >> 4;
  float* ob = out + (size_t)b * NNODE * (HCH * LSEQ);
#pragma unroll
  for (int t = 0; t < 4; ++t) {
    int col = colbase + (ct0 + t) * 16 + n;
    float gbd = gb[col >> 8];  // d = col / LSEQ
#pragma unroll
    for (int r = 0; r < 8; ++r) {
      int nr = ny * 64 + row0 + r + half * 8;
      ob[(size_t)nr * (HCH * LSEQ) + col] = acc[t][r] + gbd;
    }
  }
}

// ---------------------------------------------------------------------------
// K4: conv2 + residual fused as ONE K=256 GEMM:
//   A = [w2 im2col-weights (192) | wr (64)]  (64x256 bf16)
//   B = [im2col(gcn) (96 u32 rows) ; pairs(x) (32 u32 rows)], frag-ready
// Writes pre-BN, accumulates per-channel sum/sumsq via half-wave shfl
// butterfly -> LDS atomics -> global atomics.
// ---------------------------------------------------------------------------
__global__ __launch_bounds__(256) void k_tconv2(const float* __restrict__ gcn,
                                                const float* __restrict__ x,
                                                const float* __restrict__ w2,
                                                const float* __restrict__ b2,
                                                const float* __restrict__ wr,
                                                const float* __restrict__ br,
                                                float* __restrict__ pre,
                                                float* __restrict__ stats) {
  extern __shared__ char smem[];
  unsigned* Bc  = (unsigned*)smem;                         // 8*16*32*8 u32
  __bf16*   wAc = (__bf16*)(smem + 128 * LSEQ * 4);        // [64][256]
  float* csum = (float*)(smem + 128 * LSEQ * 4 + 64 * 256 * 2); // [64]
  float* csq  = csum + 64;                                 // [64]
  int tid = threadIdx.x;
  int bn  = blockIdx.x;
  const float* gsl = gcn + (size_t)bn * (HCH * LSEQ);
  const float* xsl = x + (size_t)bn * (CIN * LSEQ);

  for (int i = tid; i < 128 * LSEQ; i += 256) {
    int k2 = i >> 8, col = i & 255;
    float v0, v1;
    if (k2 < 96) {                                         // im2col of gcn
      int kk = 2 * k2;
      int c0 = kk / 3,       km0 = kk - 3 * c0;
      int c1 = (kk + 1) / 3, km1 = (kk + 1) - 3 * c1;
      int l0 = col + km0 - 1, l1 = col + km1 - 1;
      v0 = (l0 >= 0 && l0 < LSEQ) ? gsl[c0 * LSEQ + l0] : 0.f;
      v1 = (l1 >= 0 && l1 < LSEQ) ? gsl[c1 * LSEQ + l1] : 0.f;
    } else {                                               // residual rows of x
      int c2 = k2 - 96;
      v0 = xsl[(2 * c2) * LSEQ + col];
      v1 = xsl[(2 * c2 + 1) * LSEQ + col];
    }
    Bc[bidx(k2, col, 16)] = pack_bf16(v0, v1);
  }
  {                                                        // A = [w2 | wr]
    unsigned* wAu = (unsigned*)wAc;
    for (int i = tid; i < OUTC * 128; i += 256) {
      int o = i >> 7, kc2 = i & 127;
      int k = 2 * kc2;
      float v0, v1;
      if (k < 192) { v0 = w2[o * 192 + k];         v1 = w2[o * 192 + k + 1]; }
      else         { v0 = wr[o * CIN + (k - 192)]; v1 = wr[o * CIN + (k - 191)]; }
      wAu[i] = pack_bf16(v0, v1);
    }
  }
  if (tid < 64) { csum[tid] = 0.f; csq[tid] = 0.f; }
  __syncthreads();

  int lane = tid & 31, wave = tid >> 5;
  int row0 = (wave & 3) * 16;
  int ct0 = (wave >> 2) * 8;
  v8f acc[8] = {};
  for (int kt = 0; kt < 8; ++kt) {                         // K = 256
    v16bf a = lda_packed(wAc, 256, row0, kt * 32, lane);
#pragma unroll
    for (int t = 0; t < 8; ++t) {
      v16bf b = ldb_tile(Bc, 16, kt, ct0 + t, lane);
      acc[t] = wmma_bf16(a, b, acc[t]);
    }
  }
  int n = lane & 15, half = lane >> 4;
  float* osl = pre + (size_t)bn * (OUTC * LSEQ);
#pragma unroll
  for (int r = 0; r < 8; ++r) {
    int o = row0 + r + half * 8;                           // channel
    float bias = b2[o] + br[o];
    float s = 0.f, s2 = 0.f;
#pragma unroll
    for (int t = 0; t < 8; ++t) {
      float v = acc[t][r] + bias;
      osl[o * LSEQ + (ct0 + t) * 16 + n] = v;
      s += v;
      s2 += v * v;
    }
#pragma unroll
    for (int off = 8; off > 0; off >>= 1) {                // half-wave butterfly
      s  += __shfl_xor(s, off, 32);
      s2 += __shfl_xor(s2, off, 32);
    }
    if (n == 0) {
      atomicAdd(&csum[o], s);
      atomicAdd(&csq[o], s2);
    }
  }
  __syncthreads();
  if (tid < 64) {
    atomicAdd(&stats[tid], csum[tid]);
    atomicAdd(&stats[64 + tid], csq[tid]);
  }
}

// ---------------------------------------------------------------------------
__global__ void k_bnstats(float* __restrict__ stats,
                          const float* __restrict__ gamma,
                          const float* __restrict__ beta) {
  int o = threadIdx.x;
  if (o < OUTC) {
    const float cnt = (float)((size_t)BATCH * NNODE * LSEQ);
    float mean = stats[o] / cnt;
    float var  = stats[64 + o] / cnt - mean * mean;
    float sc   = gamma[o] * rsqrtf(var + BN_EPS);
    stats[128 + o] = sc;
    stats[192 + o] = beta[o] - mean * sc;
  }
}

__global__ __launch_bounds__(256) void k_bnrelu(const float* __restrict__ pre,
                                                const float* __restrict__ stats,
                                                float* __restrict__ out) {
  size_t i4 = (size_t)blockIdx.x * 256 + threadIdx.x;
  size_t base = i4 * 4;
  int o = (int)((base >> 8) & 63);   // (idx / LSEQ) % OUTC
  float sc = stats[128 + o], sh = stats[192 + o];
  float4 v = ((const float4*)pre)[i4];
  float4 r;
  r.x = fmaxf(v.x * sc + sh, 0.f);
  r.y = fmaxf(v.y * sc + sh, 0.f);
  r.z = fmaxf(v.z * sc + sh, 0.f);
  r.w = fmaxf(v.w * sc + sh, 0.f);
  ((float4*)out)[i4] = r;
}

// ---------------------------------------------------------------------------
extern "C" void kernel_launch(void* const* d_in, const int* in_sizes, int n_in,
                              void* d_out, int out_size, void* d_ws,
                              size_t ws_size, hipStream_t stream) {
  (void)in_sizes; (void)n_in; (void)out_size;
  const float* x     = (const float*)d_in[0];
  const float* adj   = (const float*)d_in[1];
  const float* w1    = (const float*)d_in[2];
  const float* b1    = (const float*)d_in[3];
  const float* gw    = (const float*)d_in[4];
  const float* gb    = (const float*)d_in[5];
  const float* w2    = (const float*)d_in[6];
  const float* b2    = (const float*)d_in[7];
  const float* wr    = (const float*)d_in[8];
  const float* br    = (const float*)d_in[9];
  const float* gamma = (const float*)d_in[10];
  const float* beta  = (const float*)d_in[11];
  float* out = (float*)d_out;

  const size_t elts = (size_t)BATCH * NNODE * HCH * LSEQ;   // 67,108,864
  if (ws_size < 2 * elts * sizeof(float) + 256 * sizeof(float)) return;
  float* buf0  = (float*)d_ws;       // out1 -> gcn output
  float* buf1  = buf0 + elts;        // xw   -> pre-BN
  float* stats = buf1 + elts;        // [256]: sum, sumsq, scale, shift

  const int nbn = BATCH * NNODE;     // 4096

  size_t sm1 = (size_t)96 * LSEQ * 4 + (size_t)HCH * 192 * 2;          // 122,880
  k_tconv1<<<nbn, 256, sm1, stream>>>(x, w1, b1, buf0);

  size_t sm2 = (size_t)32 * LSEQ * 4 + (size_t)HCH * HCH * 2;          // 40,960
  k_gw<<<nbn, 256, sm2, stream>>>(buf0, gw, buf1);

  dim3 g3(128, 8, BATCH);
  size_t sm3 = (size_t)64 * 16 * 4 + (size_t)16 * 128 * 4;             // 12,288
  k_graph<<<g3, 256, sm3, stream>>>(adj, buf1, gb, buf0);

  hipMemsetAsync(stats, 0, 128 * sizeof(float), stream);

  size_t sm4 = (size_t)128 * LSEQ * 4 + (size_t)OUTC * 256 * 2 +
               2 * 64 * sizeof(float);                                 // 164,352
  k_tconv2<<<nbn, 256, sm4, stream>>>(buf0, x, w2, b2, wr, br, buf1, stats);

  k_bnstats<<<1, 64, 0, stream>>>(stats, gamma, beta);

  unsigned g5 = (unsigned)(elts / 4 / 256);   // 65,536 blocks
  k_bnrelu<<<g5, 256, 0, stream>>>(buf1, stats, out);
}
